// GINet_4355096838271
// MI455X (gfx1250) — matrix operand order repaired and checked by reading, not compiled, not run on hardware
//
#include <hip/hip_runtime.h>

// ---------------- problem constants (fixed by the reference) ----------------
#define NN   50000
#define EE   600000
#define GG   512
#define DD   128
#define EDD  16
#define LL   3
#define JKD  384   // L*D
#define L1D  256
#define L2D  128

typedef __attribute__((ext_vector_type(16))) __bf16 v16bf;
typedef __attribute__((ext_vector_type(8)))  float  v8f;
typedef int v4i __attribute__((vector_size(16)));   // matches async-LDS builtin param

#if defined(__has_builtin)
#if __has_builtin(__builtin_amdgcn_global_load_async_to_lds_b128)
#define HAVE_ASYNC_LDS 1
#endif
#endif

#define AS1 __attribute__((address_space(1)))
#define AS3 __attribute__((address_space(3)))

// ---------------------------------------------------------------------------
// WMMA fragment swizzles (CDNA5 ISA §7.12.2, wave32, 16-bit data).
// A 16xK tile: lane = (r&15) + 16*half, VGPR v -> K = 16*(v>>2)+8*half+2*(v&3)
//   element (r,k) -> flat ((k>>5)*32 + lane)*16 + j   (one v16bf per lane/chunk)
// B KxN: lane = (n&15) + 16*half(=k range), VGPR v -> rows 16*(v>>2)+8*half+2*(v&3)(+1)
//   element (k,n) -> flat (((n>>4)*(K/32) + (k>>5))*32 + lane)*16 + j
// ---------------------------------------------------------------------------
__device__ __forceinline__ int swz_a(int r, int k) {       // within 16 x K tile
    int kk   = k & 31;
    int half = (kk >> 3) & 1;
    int v    = (((kk >> 4) & 1) << 2) | ((kk >> 1) & 3);
    int lane = (r & 15) + (half << 4);
    int j    = (v << 1) | (kk & 1);
    return (((k >> 5) << 5) + lane) * 16 + j;
}
__device__ __forceinline__ int swz_b(int k, int n, int K) { // within K x N matrix
    int kk   = k & 31;
    int half = (kk >> 3) & 1;
    int v    = (((kk >> 4) & 1) << 2) | ((kk >> 1) & 3);
    int lane = (n & 15) + (half << 4);
    int j    = (v << 1) | (kk & 1);
    return (((n >> 4) * (K >> 5) + (k >> 5)) * 32 + lane) * 16 + j;
}

// ---------------------------------------------------------------------------
// Batched f32 -> bf16 weight conversion into B-fragment-swizzled layout.
// ---------------------------------------------------------------------------
__global__ void cvt_swz_b_kernel(const float* __restrict__ in, __bf16* __restrict__ out,
                                 int K, int N, int total) {
    int i = blockIdx.x * blockDim.x + threadIdx.x;
    if (i >= total) return;
    int kn  = K * N;
    int mat = i / kn, rem = i - mat * kn;
    int k = rem / N, n = rem - k * N;
    out[(size_t)mat * kn + swz_b(k, n, K)] = (__bf16)in[i];
}

// f32 [M,K] -> bf16 A-fragment-swizzled (row tiles of 16)
__global__ void cvt_swz_a_kernel(const float* __restrict__ in, __bf16* __restrict__ out,
                                 int M, int K) {
    int i = blockIdx.x * blockDim.x + threadIdx.x;
    if (i >= M * K) return;
    int r = i / K, k = i - r * K;
    out[(size_t)(r >> 4) * (16 * K) + swz_a(r, k)] = (__bf16)in[i];
}

// ---------------------------------------------------------------------------
// Fused edge message: m = relu((x[src]+vn) + edge_attr@We + be); atomicAdd agg[dst]
// vn broadcast folded in (no materialized x+vn pass). One wave per edge;
// lane l covers dims [4l..4l+3]. We (16x128 f32) in LDS; gather/scatter hit L2.
// ---------------------------------------------------------------------------
__global__ void edge_msg_kernel(const int* __restrict__ src, const int* __restrict__ dst,
                                const float* __restrict__ eattr,
                                const float* __restrict__ We, const float* __restrict__ be,
                                const float* __restrict__ xin, int stride, int colOff,
                                const float* __restrict__ vn, float* __restrict__ agg) {
    __shared__ float sW[EDD * DD];
    __shared__ float sB[DD];
    int t = threadIdx.x;
    for (int j = t; j < EDD * DD; j += 256) sW[j] = We[j];
    if (t < DD) sB[t] = be[t];
    __syncthreads();

    int lane = t & 31;
    int wave = t >> 5;
    int gw = blockIdx.x * 8 + wave;
    int nw = gridDim.x * 8;
    int d0 = lane * 4;
    float4 vn4 = *(const float4*)(vn + d0);

    for (int e = gw; e < EE; e += nw) {
        int s = src[e];
        int d = dst[e];
        const float* ea = eattr + (size_t)e * EDD;
        __builtin_prefetch(eattr + (size_t)(e + nw) * EDD, 0, 1); // global_prefetch_b8
        float4 xv = *(const float4*)(xin + (size_t)s * stride + colOff + d0);
        float m0 = sB[d0 + 0] + vn4.x, m1 = sB[d0 + 1] + vn4.y;
        float m2 = sB[d0 + 2] + vn4.z, m3 = sB[d0 + 3] + vn4.w;
#pragma unroll
        for (int k = 0; k < EDD; ++k) {
            float a = ea[k];
            const float* wr = sW + k * DD + d0;
            m0 = fmaf(a, wr[0], m0);
            m1 = fmaf(a, wr[1], m1);
            m2 = fmaf(a, wr[2], m2);
            m3 = fmaf(a, wr[3], m3);
        }
        m0 += xv.x; m1 += xv.y; m2 += xv.z; m3 += xv.w;
        m0 = m0 > 0.f ? m0 : 0.f;  m1 = m1 > 0.f ? m1 : 0.f;
        m2 = m2 > 0.f ? m2 : 0.f;  m3 = m3 > 0.f ? m3 : 0.f;
        float* ap = agg + (size_t)d * DD + d0;
        atomicAdd(ap + 0, m0); atomicAdd(ap + 1, m1);
        atomicAdd(ap + 2, m2); atomicAdd(ap + 3, m3);
    }
}

// ---------------------------------------------------------------------------
// GIN node MLP: h=(1+eps)*(x+vn)+agg; out = relu(h@W1+b1)@W2+b2 -> xj[:,layer*128]
// One block = 16 nodes, 8 waves; wave w owns output cols [16w,16w+16).
// W1/W2 double-buffered in LDS, fetched with GLOBAL_LOAD_ASYNC_TO_LDS_B128 at
// kernel entry (overlaps h-tile build + GEMM1); ASYNCcnt split-waits gate use.
// Every fragment is one aligned 32B LDS load (2x ds_load_b128). 8 wmma/wave.
// ---------------------------------------------------------------------------
__global__ void node_mlp_kernel(const float* __restrict__ xin, int stride, int colOff,
                                const float* __restrict__ vn,
                                const float* __restrict__ agg,
                                const float* __restrict__ epsArr, int layer,
                                const __bf16* __restrict__ W1, const float* __restrict__ b1,
                                const __bf16* __restrict__ W2, const float* __restrict__ b2,
                                float* __restrict__ xj) {
    __shared__ v16bf sW1[1024];    // 32KB, W1 swizzled
    __shared__ v16bf sW2[1024];    // 32KB, W2 swizzled
    __shared__ v16bf sH[128];      // 16x128 h tile, A-fragment order
    __shared__ v16bf sA[128];      // 16x128 relu(h@W1+b1) tile, A-fragment order

    int t = threadIdx.x;
    int n0 = blockIdx.x * 16;
    float ep = 1.0f + epsArr[layer];

#if HAVE_ASYNC_LDS
    {
        AS1 char* g1 = (AS1 char*)W1;
        AS1 char* g2 = (AS1 char*)W2;
        AS3 char* l1 = (AS3 char*)sW1;
        AS3 char* l2 = (AS3 char*)sW2;
        int byteOff = t * 16;
#pragma unroll
        for (int j = 0; j < 8; ++j) {        // 8 x 4KB = 32KB (W1 first)
            __builtin_amdgcn_global_load_async_to_lds_b128(
                (AS1 v4i*)(g1 + byteOff), (AS3 v4i*)(l1 + byteOff), 0, 0);
            byteOff += 256 * 16;
        }
        byteOff = t * 16;
#pragma unroll
        for (int j = 0; j < 8; ++j) {        // W2 second (retires after W1)
            __builtin_amdgcn_global_load_async_to_lds_b128(
                (AS1 v4i*)(g2 + byteOff), (AS3 v4i*)(l2 + byteOff), 0, 0);
            byteOff += 256 * 16;
        }
    }
#else
    {
        uint4* s1 = (uint4*)sW1;
        uint4* s2 = (uint4*)sW2;
        const uint4* g1 = (const uint4*)W1;
        const uint4* g2 = (const uint4*)W2;
        for (int j = t; j < 2048; j += 256) { s1[j] = g1[j]; s2[j] = g2[j]; }
    }
#endif

    // build h tile (overlaps the async weight DMA)
    __bf16* sHe = (__bf16*)sH;
#pragma unroll
    for (int j = t; j < 16 * DD; j += 256) {
        int r = j >> 7, d = j & 127;
        float xv = xin[(size_t)(n0 + r) * stride + colOff + d] + vn[d];
        float hv = ep * xv + agg[(size_t)(n0 + r) * DD + d];
        sHe[swz_a(r, d)] = (__bf16)hv;
    }
#if HAVE_ASYNC_LDS
    asm volatile("s_wait_asynccnt 0x8" ::: "memory");   // W1 half complete
#endif
    __syncthreads();

    int lane = t & 31;
    int w    = t >> 5;          // output col tile 0..7
    int m    = lane & 15;
    int half = lane >> 4;

    // ---- GEMM1: h @ W1 ----
    v8f acc = {};
#pragma unroll
    for (int kc = 0; kc < 4; ++kc) {
        v16bf a = sH[kc * 32 + lane];
        v16bf b = sW1[(w * 4 + kc) * 32 + lane];
        acc = __builtin_amdgcn_wmma_f32_16x16x32_bf16(false, a, false, b,
                                                      (short)0, acc, false, false);
    }
    __bf16* sAe = (__bf16*)sA;
#pragma unroll
    for (int v = 0; v < 8; ++v) {
        int Mr  = v + 8 * half;
        int col = w * 16 + m;
        float val = acc[v] + b1[col];
        sAe[swz_a(Mr, col)] = (__bf16)(val > 0.f ? val : 0.f);
    }
#if HAVE_ASYNC_LDS
    asm volatile("s_wait_asynccnt 0x0" ::: "memory");   // W2 complete
#endif
    __syncthreads();                                    // sA + all W2 visible

    // ---- GEMM2: a1 @ W2 ----
    v8f acc2 = {};
#pragma unroll
    for (int kc = 0; kc < 4; ++kc) {
        v16bf a = sA[kc * 32 + lane];
        v16bf b = sW2[(w * 4 + kc) * 32 + lane];
        acc2 = __builtin_amdgcn_wmma_f32_16x16x32_bf16(false, a, false, b,
                                                       (short)0, acc2, false, false);
    }
#pragma unroll
    for (int v = 0; v < 8; ++v) {
        int Mr  = v + 8 * half;
        int col = w * 16 + m;
        xj[(size_t)(n0 + Mr) * JKD + layer * DD + col] = acc2[v] + b2[col];
    }
}

// ---------------------------------------------------------------------------
// global_add_pool: g[batch[n], :] += xj[n, :]  (batch sorted; atomics hit L2)
// ---------------------------------------------------------------------------
__global__ void pool_kernel(const float* __restrict__ xj, const int* __restrict__ batch,
                            float* __restrict__ g) {
    int idx = blockIdx.x * blockDim.x + threadIdx.x;   // N * 96 float4 chunks
    if (idx >= NN * (JKD / 4)) return;
    int n = idx / (JKD / 4);
    int c = (idx % (JKD / 4)) * 4;
    int b = batch[n];
    float4 v = *(const float4*)(xj + (size_t)n * JKD + c);
    float* gp = g + (size_t)b * JKD + c;
    atomicAdd(gp + 0, v.x); atomicAdd(gp + 1, v.y);
    atomicAdd(gp + 2, v.z); atomicAdd(gp + 3, v.w);
}

// ---------------------------------------------------------------------------
// Head GEMM: C[M,N] = A @ B + bias, A and B both bf16 pre-swizzled in global.
// One wave per 16x16 tile; each fragment = one 32B global load pair.
// ---------------------------------------------------------------------------
__global__ void head_gemm_kernel(const v16bf* __restrict__ As, const v16bf* __restrict__ Bs,
                                 const float* __restrict__ bias, float* __restrict__ C,
                                 int M, int N, int K) {
    int ntiles = N >> 4;
    int tr = blockIdx.x / ntiles;
    int tc = blockIdx.x % ntiles;
    int lane = threadIdx.x & 31;
    int m = lane & 15, half = lane >> 4;

    v8f acc = {};
    for (int kc = 0; kc < (K >> 5); ++kc) {
        v16bf a = As[(size_t)tr * K + kc * 32 + lane];            // 16*K elems / tile
        v16bf b = Bs[((size_t)tc * (K >> 5) + kc) * 32 + lane];
        acc = __builtin_amdgcn_wmma_f32_16x16x32_bf16(false, a, false, b,
                                                      (short)0, acc, false, false);
    }
#pragma unroll
    for (int v = 0; v < 8; ++v) {
        int Mr  = v + 8 * half;
        int col = tc * 16 + m;
        C[(size_t)(tr * 16 + Mr) * N + col] = acc[v] + bias[col];
    }
}

// ---------------------------------------------------------------------------
// Row LayerNorm + ReLU, one wave per row (shuffle reductions, wave32)
// ---------------------------------------------------------------------------
__global__ void ln_relu_kernel(const float* __restrict__ in, const float* __restrict__ gam,
                               const float* __restrict__ bet, float* __restrict__ out,
                               int cols) {
    int row  = (blockIdx.x * blockDim.x + threadIdx.x) >> 5;
    int lane = threadIdx.x & 31;
    const float* r = in + (size_t)row * cols;
    float s = 0.f, ss = 0.f;
    for (int c = lane; c < cols; c += 32) { float v = r[c]; s += v; ss += v * v; }
#pragma unroll
    for (int o = 16; o; o >>= 1) { s += __shfl_xor(s, o, 32); ss += __shfl_xor(ss, o, 32); }
    float mu  = s / cols;
    float var = ss / cols - mu * mu;
    float inv = rsqrtf(var + 1e-5f);
    float* orow = out + (size_t)row * cols;
    for (int c = lane; c < cols; c += 32) {
        float v = (r[c] - mu) * inv * gam[c] + bet[c];
        orow[c] = v > 0.f ? v : 0.f;
    }
}

// ---------------------------------------------------------------------------
// Final projection: out[g] = dot(in[g,:], out_w) + out_b ; one wave per graph
// ---------------------------------------------------------------------------
__global__ void head_out_kernel(const float* __restrict__ in, const float* __restrict__ ow,
                                const float* __restrict__ ob, float* __restrict__ out) {
    int row  = (blockIdx.x * blockDim.x + threadIdx.x) >> 5;
    int lane = threadIdx.x & 31;
    const float* r = in + (size_t)row * L2D;
    float s = 0.f;
    for (int c = lane; c < L2D; c += 32) s = fmaf(r[c], ow[c], s);
#pragma unroll
    for (int o = 16; o; o >>= 1) s += __shfl_xor(s, o, 32);
    if (lane == 0) out[row] = s + ob[0];
}

// ---------------------------------------------------------------------------
extern "C" void kernel_launch(void* const* d_in, const int* in_sizes, int n_in,
                              void* d_out, int out_size, void* d_ws, size_t ws_size,
                              hipStream_t stream) {
    const float* x      = (const float*)d_in[0];
    const int*   eidx   = (const int*)  d_in[1];
    const float* eattr  = (const float*)d_in[2];
    const int*   batch  = (const int*)  d_in[3];
    const float* vn_emb = (const float*)d_in[4];
    const float* edge_w = (const float*)d_in[5];
    const float* edge_b = (const float*)d_in[6];
    const float* epsArr = (const float*)d_in[7];
    const float* mlp_w1 = (const float*)d_in[8];
    const float* mlp_b1 = (const float*)d_in[9];
    const float* mlp_w2 = (const float*)d_in[10];
    const float* mlp_b2 = (const float*)d_in[11];
    const float* lin_w1 = (const float*)d_in[12];
    const float* lin_b1 = (const float*)d_in[13];
    const float* ln1_g  = (const float*)d_in[14];
    const float* ln1_b  = (const float*)d_in[15];
    const float* lin_w2 = (const float*)d_in[16];
    const float* lin_b2 = (const float*)d_in[17];
    const float* ln2_g  = (const float*)d_in[18];
    const float* ln2_b  = (const float*)d_in[19];
    const float* out_w  = (const float*)d_in[20];
    const float* out_b  = (const float*)d_in[21];
    float* out = (float*)d_out;

    // ---- workspace carve-out ----
    char*  ws  = (char*)d_ws;
    size_t off = 0;
    auto carve = [&](size_t bytes) {
        char* p = ws + off;
        off = (off + bytes + 255) & ~(size_t)255;
        return p;
    };
    float*  agg   = (float*) carve((size_t)NN * DD * 4);
    float*  xj    = (float*) carve((size_t)NN * JKD * 4);
    float*  gpool = (float*) carve((size_t)GG * JKD * 4);
    float*  t1    = (float*) carve((size_t)GG * L1D * 4);
    float*  t1n   = (float*) carve((size_t)GG * L1D * 4);
    float*  t2    = (float*) carve((size_t)GG * L2D * 4);
    float*  t2n   = (float*) carve((size_t)GG * L2D * 4);
    __bf16* w1bf  = (__bf16*)carve((size_t)LL * DD * DD * 2);   // swizzled B
    __bf16* w2bf  = (__bf16*)carve((size_t)LL * DD * DD * 2);   // swizzled B
    __bf16* lw1bf = (__bf16*)carve((size_t)JKD * L1D * 2);      // swizzled B
    __bf16* lw2bf = (__bf16*)carve((size_t)L1D * L2D * 2);      // swizzled B
    __bf16* gs    = (__bf16*)carve((size_t)GG * JKD * 2);       // swizzled A
    __bf16* t1s   = (__bf16*)carve((size_t)GG * L1D * 2);       // swizzled A
    (void)ws_size; (void)in_sizes; (void)n_in; (void)out_size;

    // ---- weight conversion to swizzled bf16 ----
    {
        int n1 = LL * DD * DD, n2 = JKD * L1D, n3 = L1D * L2D;
        cvt_swz_b_kernel<<<(n1 + 255) / 256, 256, 0, stream>>>(mlp_w1, w1bf, DD, DD, n1);
        cvt_swz_b_kernel<<<(n1 + 255) / 256, 256, 0, stream>>>(mlp_w2, w2bf, DD, DD, n1);
        cvt_swz_b_kernel<<<(n2 + 255) / 256, 256, 0, stream>>>(lin_w1, lw1bf, JKD, L1D, n2);
        cvt_swz_b_kernel<<<(n3 + 255) / 256, 256, 0, stream>>>(lin_w2, lw2bf, L1D, L2D, n3);
    }

    const int* src = eidx;
    const int* dst = eidx + EE;

    for (int i = 0; i < LL; ++i) {
        (void)hipMemsetAsync(agg, 0, (size_t)NN * DD * 4, stream);
        const float* xin  = (i == 0) ? x : xj;
        int stride = (i == 0) ? DD : JKD;
        int colOff = (i == 0) ? 0 : (i - 1) * DD;
        const float* vn = vn_emb + i * DD;
        edge_msg_kernel<<<4096, 256, 0, stream>>>(
            src, dst, eattr, edge_w + (size_t)i * EDD * DD, edge_b + i * DD,
            xin, stride, colOff, vn, agg);
        node_mlp_kernel<<<NN / 16, 256, 0, stream>>>(
            xin, stride, colOff, vn, agg, epsArr, i,
            w1bf + (size_t)i * DD * DD, mlp_b1 + i * DD,
            w2bf + (size_t)i * DD * DD, mlp_b2 + i * DD, xj);
    }

    (void)hipMemsetAsync(gpool, 0, (size_t)GG * JKD * 4, stream);
    pool_kernel<<<(NN * (JKD / 4) + 255) / 256, 256, 0, stream>>>(xj, batch, gpool);

    // head: 384 -> 256 -> 128 -> 1
    cvt_swz_a_kernel<<<(GG * JKD + 255) / 256, 256, 0, stream>>>(gpool, gs, GG, JKD);
    head_gemm_kernel<<<(GG / 16) * (L1D / 16), 32, 0, stream>>>(
        (const v16bf*)gs, (const v16bf*)lw1bf, lin_b1, t1, GG, L1D, JKD);
    ln_relu_kernel<<<GG / 8, 256, 0, stream>>>(t1, ln1_g, ln1_b, t1n, L1D);
    cvt_swz_a_kernel<<<(GG * L1D + 255) / 256, 256, 0, stream>>>(t1n, t1s, GG, L1D);
    head_gemm_kernel<<<(GG / 16) * (L2D / 16), 32, 0, stream>>>(
        (const v16bf*)t1s, (const v16bf*)lw2bf, lin_b2, t2, GG, L2D, L1D);
    ln_relu_kernel<<<GG / 8, 256, 0, stream>>>(t2, ln2_g, ln2_b, t2n, L2D);
    head_out_kernel<<<GG / 8, 256, 0, stream>>>(t2n, out_w, out_b, out);
}
